// BiSpDiff_17265768530288
// MI455X (gfx1250) — matrix operand order repaired and checked by reading, not compiled, not run on hardware
//
#include <hip/hip_runtime.h>
#include <hip/hip_bf16.h>

typedef __attribute__((ext_vector_type(16))) __bf16 v16bf;
typedef __attribute__((ext_vector_type(8)))  __bf16 v8bf;
typedef __attribute__((ext_vector_type(8)))  float  v8f;

#define NN 4096
#define FF 128

// ---------------------------------------------------------------------------
// 1) adj (f32) -> A (bf16, diag zeroed) and A^T (bf16), 32x32 LDS tiles
// ---------------------------------------------------------------------------
__global__ void __launch_bounds__(256)
convert_transpose(const float* __restrict__ adj,
                  __bf16* __restrict__ Abf, __bf16* __restrict__ Atbf) {
    __shared__ float tile[32][33];
    const int tx = threadIdx.x;          // 0..31
    const int ty = threadIdx.y;          // 0..7
    const int bx = blockIdx.x;           // col tile
    const int by = blockIdx.y;           // row tile
    int x = bx * 32 + tx;
    #pragma unroll
    for (int i = 0; i < 32; i += 8) {
        int y = by * 32 + ty + i;
        float v = adj[(size_t)y * NN + x];
        if (y == x) v = 0.0f;
        Abf[(size_t)y * NN + x] = (__bf16)v;
        tile[ty + i][tx] = v;
    }
    __syncthreads();
    x = by * 32 + tx;                    // swapped tile origin
    #pragma unroll
    for (int i = 0; i < 32; i += 8) {
        int y = bx * 32 + ty + i;
        Atbf[(size_t)y * NN + x] = (__bf16)tile[tx][ty + i];
    }
}

// ---------------------------------------------------------------------------
// 2) degree sums.  rowsum: block/row with LDS reduce (diag excluded here).
//    colsum: 16 row-chunks x 4096 cols, f32 global atomicAdd partials
//    (diag excluded later in deg_inv_k).
// ---------------------------------------------------------------------------
__global__ void __launch_bounds__(256)
row_sum(const float* __restrict__ adj, float* __restrict__ rs) {
    __shared__ float sm[256];
    const int row = blockIdx.x;
    const float* r = adj + (size_t)row * NN;
    float s = 0.0f;
    for (int c = threadIdx.x; c < NN; c += 256) s += r[c];
    sm[threadIdx.x] = s;
    __syncthreads();
    for (int off = 128; off > 0; off >>= 1) {
        if (threadIdx.x < off) sm[threadIdx.x] += sm[threadIdx.x + off];
        __syncthreads();
    }
    if (threadIdx.x == 0) rs[row] = sm[0] - r[row];
}

__global__ void __launch_bounds__(256)
zero_vec(float* __restrict__ v) {
    v[blockIdx.x * 256 + threadIdx.x] = 0.0f;
}

__global__ void __launch_bounds__(256)
col_sum_partial(const float* __restrict__ adj, float* __restrict__ cs) {
    const int c  = blockIdx.x * 256 + threadIdx.x;   // column
    const int r0 = blockIdx.y * (NN / 16);           // row chunk
    float s = 0.0f;
    for (int r = r0; r < r0 + NN / 16; ++r) s += adj[(size_t)r * NN + c];
    atomicAdd(&cs[c], s);
}

__global__ void __launch_bounds__(256)
deg_inv_k(const float* __restrict__ rs, const float* __restrict__ cs,
          const float* __restrict__ adj, float* __restrict__ di) {
    const int i = blockIdx.x * 256 + threadIdx.x;
    const float d = rs[i] + cs[i] - adj[(size_t)i * NN + i];  // cs includes diag
    di[i] = (d == 0.0f) ? 0.0f : 1.0f / d;
}

// ---------------------------------------------------------------------------
// 3) pack skinny operand [K x 128] (f32 row-major) into WMMA B-fragment order:
//    fragment element e of lane l in tile (kt,nt):
//      K = kt*32 + (l>=16 ? 8 : 0) + (e<8 ? e : e+8),  N = nt*16 + (l&15)
//    linear dst index = (((kt*8 + nt)*32 + lane)*16 + e)
// ---------------------------------------------------------------------------
__global__ void __launch_bounds__(256)
pack_b(const float* __restrict__ src, __bf16* __restrict__ dst) {
    const size_t idx = (size_t)blockIdx.x * 256 + threadIdx.x;  // K*128 elems
    const int e    = (int)(idx & 15);
    const int lane = (int)((idx >> 4) & 31);
    const size_t tile = idx >> 9;
    const int nt = (int)(tile & 7);
    const int kt = (int)(tile >> 3);
    const int k = kt * 32 + ((lane & 16) ? 8 : 0) + (e < 8 ? e : e + 8);
    const int n = nt * 16 + (lane & 15);
    dst[idx] = (__bf16)src[(size_t)k * FF + n];
}

// pack W [128 out x 128 in] as B = W^T:  B[k][n] = W[n][k]
__global__ void __launch_bounds__(256)
pack_w(const float* __restrict__ w, __bf16* __restrict__ dst) {
    const size_t idx = (size_t)blockIdx.x * 256 + threadIdx.x;  // 128*128
    const int e    = (int)(idx & 15);
    const int lane = (int)((idx >> 4) & 31);
    const size_t tile = idx >> 9;
    const int nt = (int)(tile & 7);
    const int kt = (int)(tile >> 3);
    const int k = kt * 32 + ((lane & 16) ? 8 : 0) + (e < 8 ? e : e + 8);
    const int n = nt * 16 + (lane & 15);
    dst[idx] = (__bf16)w[(size_t)n * FF + k];
}

// ---------------------------------------------------------------------------
// 4) C[4096x128] = diag(scale) * (A[4096x4096,bf16] @ Bpk)   (f32 out)
//    1 wave = 16x32 output tile (2 accumulators); grid 64 x 4 = 1024 waves.
//    A is L2-resident (32MB << 192MB L2), so the 4x A re-read is cheap;
//    the extra waves hide the load->wmma wait chains.
// ---------------------------------------------------------------------------
__global__ void __launch_bounds__(128)
gemm_scaled(const __bf16* __restrict__ A, const __bf16* __restrict__ Bpk,
            const float* __restrict__ scale, float* __restrict__ C) {
    const int lane = threadIdx.x & 31;
    const int wave = threadIdx.x >> 5;
    const int m0   = (blockIdx.x * 4 + wave) * 16;
    const int nt0  = blockIdx.y * 2;        // this wave's two n-tiles
    const int lm   = lane & 15;
    const int half = (lane & 16) ? 8 : 0;   // K-offset for A frag, M-offset for C

    v8f acc0 = {}, acc1 = {};
    const __bf16* abase = A + (size_t)(m0 + lm) * NN + half;
    const __bf16* bbase = Bpk + ((size_t)nt0 * 32 + lane) * 16;

    for (int kt = 0; kt < NN / 32; ++kt) {
        const __bf16* ap = abase + kt * 32;
        __builtin_prefetch(ap + 512, 0, 0);
        v8bf lo = *(const v8bf*)(ap);
        v8bf hi = *(const v8bf*)(ap + 16);
        v16bf afrag = __builtin_shufflevector(lo, hi,
            0, 1, 2, 3, 4, 5, 6, 7, 8, 9, 10, 11, 12, 13, 14, 15);
        const __bf16* bp = bbase + (size_t)kt * (8 * 32 * 16);
        v16bf b0 = *(const v16bf*)(bp);
        v16bf b1 = *(const v16bf*)(bp + 32 * 16);
        acc0 = __builtin_amdgcn_wmma_f32_16x16x32_bf16(
            false, afrag, false, b0, (short)0, acc0, false, false);
        acc1 = __builtin_amdgcn_wmma_f32_16x16x32_bf16(
            false, afrag, false, b1, (short)0, acc1, false, false);
    }
    #pragma unroll
    for (int j = 0; j < 8; ++j) {
        const int m = m0 + j + half;
        const float s = scale[m];
        float* crow = C + (size_t)m * FF + lm;
        crow[nt0 * 16]       = acc0[j] * s;
        crow[(nt0 + 1) * 16] = acc1[j] * s;
    }
}

// ---------------------------------------------------------------------------
// 5) aggr = 0.5*y1 + 0.25*y2  ->  bf16 row-major (A operand of final GEMM)
// ---------------------------------------------------------------------------
__global__ void __launch_bounds__(256)
combine(const float* __restrict__ y1, const float* __restrict__ y2,
        __bf16* __restrict__ out) {
    const size_t i = (size_t)blockIdx.x * 256 + threadIdx.x;
    out[i] = (__bf16)(0.5f * y1[i] + 0.25f * y2[i]);
}

// ---------------------------------------------------------------------------
// 6) out = relu(agg1 @ W1^T + b1) + relu(agg2 @ W2^T + b2)
//    1 wave = one 16x16 output tile; grid 64 x 8 = 2048 waves.
// ---------------------------------------------------------------------------
__global__ void __launch_bounds__(128)
final_gemm(const __bf16* __restrict__ agg1, const __bf16* __restrict__ agg2,
           const __bf16* __restrict__ w1pk, const __bf16* __restrict__ w2pk,
           const float* __restrict__ b1, const float* __restrict__ b2,
           float* __restrict__ out) {
    const int lane = threadIdx.x & 31;
    const int wave = threadIdx.x >> 5;
    const int m0   = (blockIdx.x * 4 + wave) * 16;
    const int nt   = blockIdx.y;             // 0..7
    const int lm   = lane & 15;
    const int half = (lane & 16) ? 8 : 0;

    const __bf16* aggs[2] = {agg1, agg2};
    const __bf16* wpks[2] = {w1pk, w2pk};
    const float*  bias[2] = {b1, b2};

    v8f res = {};
    #pragma unroll
    for (int br = 0; br < 2; ++br) {
        v8f acc = {};
        const __bf16* abase = aggs[br] + (size_t)(m0 + lm) * FF + half;
        #pragma unroll
        for (int kt = 0; kt < 4; ++kt) {
            const __bf16* ap = abase + kt * 32;
            v8bf lo = *(const v8bf*)(ap);
            v8bf hi = *(const v8bf*)(ap + 16);
            v16bf afrag = __builtin_shufflevector(lo, hi,
                0, 1, 2, 3, 4, 5, 6, 7, 8, 9, 10, 11, 12, 13, 14, 15);
            v16bf bfrag = *(const v16bf*)(wpks[br] +
                ((size_t)(kt * 8 + nt) * 32 + lane) * 16);
            acc = __builtin_amdgcn_wmma_f32_16x16x32_bf16(
                false, afrag, false, bfrag, (short)0, acc, false, false);
        }
        const float bb = bias[br][nt * 16 + lm];
        #pragma unroll
        for (int j = 0; j < 8; ++j)
            res[j] += fmaxf(acc[j] + bb, 0.0f);
    }
    #pragma unroll
    for (int j = 0; j < 8; ++j)
        out[(size_t)(m0 + j + half) * FF + nt * 16 + lm] = res[j];
}

// ---------------------------------------------------------------------------
extern "C" void kernel_launch(void* const* d_in, const int* in_sizes, int n_in,
                              void* d_out, int out_size, void* d_ws, size_t ws_size,
                              hipStream_t stream) {
    const float* x    = (const float*)d_in[0];   // [4096,128]
    const float* adj  = (const float*)d_in[1];   // [4096,4096]
    const float* W1   = (const float*)d_in[2];   // [128,128]
    const float* b1   = (const float*)d_in[3];   // [128]
    const float* W2   = (const float*)d_in[4];   // [128,128]
    const float* b2   = (const float*)d_in[5];   // [128]
    float* out        = (float*)d_out;           // [4096,128]

    char* w = (char*)d_ws;
    const size_t ABYTES = (size_t)NN * NN * 2;            // 32 MB
    __bf16* Abf   = (__bf16*)(w);
    __bf16* Atbf  = (__bf16*)(w + ABYTES);
    float*  rs    = (float*)(w + 2 * ABYTES);
    float*  cs    = rs + NN;
    float*  di    = cs + NN;
    char*   p     = w + 2 * ABYTES + 3 * NN * sizeof(float);
    const size_t PKB = (size_t)NN * FF * 2;               // 1 MB
    const size_t YB  = (size_t)NN * FF * 4;               // 2 MB
    __bf16* xpk  = (__bf16*)(p);              p += PKB;
    __bf16* ypk  = (__bf16*)(p);              p += PKB;
    float*  y1   = (float*)(p);               p += YB;
    float*  y2   = (float*)(p);               p += YB;
    float*  z1   = (float*)(p);               p += YB;
    float*  z2   = (float*)(p);               p += YB;
    __bf16* agg1 = (__bf16*)(p);              p += PKB;
    __bf16* agg2 = (__bf16*)(p);              p += PKB;
    __bf16* w1pk = (__bf16*)(p);              p += (size_t)FF * FF * 2;
    __bf16* w2pk = (__bf16*)(p);

    // --- preprocessing ---
    convert_transpose<<<dim3(NN / 32, NN / 32), dim3(32, 8), 0, stream>>>(adj, Abf, Atbf);
    row_sum<<<NN, 256, 0, stream>>>(adj, rs);
    zero_vec<<<NN / 256, 256, 0, stream>>>(cs);
    col_sum_partial<<<dim3(NN / 256, 16), 256, 0, stream>>>(adj, cs);
    deg_inv_k<<<NN / 256, 256, 0, stream>>>(rs, cs, adj, di);
    pack_b<<<(NN * FF) / 256, 256, 0, stream>>>(x, xpk);
    pack_w<<<(FF * FF) / 256, 256, 0, stream>>>(W1, w1pk);
    pack_w<<<(FF * FF) / 256, 256, 0, stream>>>(W2, w2pk);

    const dim3 gg(NN / 16 / 4, 4), gb(128);
    // forward: y1 = P x ; y2 = P y1
    gemm_scaled<<<gg, gb, 0, stream>>>(Abf, xpk, di, y1);
    pack_b<<<(NN * FF) / 256, 256, 0, stream>>>(y1, ypk);
    gemm_scaled<<<gg, gb, 0, stream>>>(Abf, ypk, di, y2);
    // reverse direction: transposed A, same degree vector
    gemm_scaled<<<gg, gb, 0, stream>>>(Atbf, xpk, di, z1);
    pack_b<<<(NN * FF) / 256, 256, 0, stream>>>(z1, ypk);
    gemm_scaled<<<gg, gb, 0, stream>>>(Atbf, ypk, di, z2);

    combine<<<(NN * FF) / 256, 256, 0, stream>>>(y1, y2, agg1);
    combine<<<(NN * FF) / 256, 256, 0, stream>>>(z1, z2, agg2);

    final_gemm<<<dim3(NN / 16 / 4, 8), gb, 0, stream>>>(agg1, agg2, w1pk, w2pk, b1, b2, out);
}
